// _GraphormerAttention_48455821033930
// MI455X (gfx1250) — matrix operand order repaired and checked by reading, not compiled, not run on hardware
//
#include <hip/hip_runtime.h>
#include <math.h>

#define BB   8
#define LL   1024
#define HIDD 1024
#define NHH  16
#define HDD  64
#define PSTR 136   // padded bf16 row stride for per-wave prob slabs (272B rows)

typedef __attribute__((ext_vector_type(16))) __bf16 v16bf;
typedef __attribute__((ext_vector_type(8)))  __bf16 v8bf;
typedef __attribute__((ext_vector_type(4)))  __bf16 v4bf;
typedef __attribute__((ext_vector_type(8)))  float  v8f;
typedef __attribute__((ext_vector_type(4)))  float  v4f;

static __device__ __forceinline__ v16bf cat8(v8bf lo, v8bf hi) {
  v16bf r;
#pragma unroll
  for (int i = 0; i < 8; ++i) { r[i] = lo[i]; r[i + 8] = hi[i]; }
  return r;
}

// A-fragment (16x32 bf16): lane m<16 holds row m, K = k0..k0+7 then k0+16..k0+23;
// lane m+16 holds K = k0+8..k0+15 then k0+24..k0+31.  rowp = row start pointer.
static __device__ __forceinline__ v16bf ldA(const __bf16* rowp, int k0, int lane) {
  const __bf16* p = rowp + k0 + ((lane >> 4) << 3);
  return cat8(*(const v8bf*)p, *(const v8bf*)(p + 16));
}

// B-fragment (32x16 bf16): lane n<16 holds col n, K = k0..k0+7 then k0+16..k0+23; lane n+16 holds K = k0+8..k0+15 then k0+24..k0+31.
static __device__ __forceinline__ v16bf ldB(const __bf16* rowp, int k0, int lane) {
  const __bf16* p = rowp + k0 + ((lane >> 4) << 3);
  return cat8(*(const v8bf*)p, *(const v8bf*)(p + 16));
}

static __device__ __forceinline__ v8f wmma_bf16(v16bf a, v16bf b, v8f c) {
  return __builtin_amdgcn_wmma_f32_16x16x32_bf16(false, a, false, b, (short)0, c, false, false);
}

// ---------------------------------------------------------------- convert f32 -> bf16
__global__ void cvt_bf16_kernel(const float* __restrict__ src, __bf16* __restrict__ dst, int n) {
  int i = (blockIdx.x * blockDim.x + threadIdx.x) * 4;
  if (i < n) {
    v4f f = *(const v4f*)(src + i);
    v4bf o;
#pragma unroll
    for (int k = 0; k < 4; ++k) o[k] = (__bf16)f[k];
    *(v4bf*)(dst + i) = o;
  }
}

// ---------------------------------------------------------------- QKV projection GEMM
// Y[m,n] = sum_k X[m,k]*W[n,k] + bias[n]; M=8192, N=1024, K=1024.
// Q,K stored [b,h,l,d] bf16; V stored transposed [b,h,d,l] bf16.
__global__ void qkv_gemm_kernel(const __bf16* __restrict__ xb,
                                const __bf16* __restrict__ wqb,
                                const __bf16* __restrict__ wkb,
                                const __bf16* __restrict__ wvb,
                                const float* __restrict__ bq,
                                const float* __restrict__ bk,
                                const float* __restrict__ bv,
                                __bf16* __restrict__ qws,
                                __bf16* __restrict__ kws,
                                __bf16* __restrict__ vtws) {
  const int lane = threadIdx.x & 31;
  const int wave = threadIdx.x >> 5;
  const int mode = blockIdx.z;  // 0=Q 1=K 2=V
  const __bf16* W    = (mode == 0) ? wqb : (mode == 1) ? wkb : wvb;
  const float*  bias = (mode == 0) ? bq  : (mode == 1) ? bk  : bv;

  const int Mbase = blockIdx.x * 64 + (wave & 1) * 32;
  const int Nbase = blockIdx.y * 64 + (wave >> 1) * 32;

  v8f acc[2][2] = {};
  const __bf16* arow0 = xb + (size_t)(Mbase + (lane & 15)) * HIDD;
  const __bf16* arow1 = arow0 + 16 * HIDD;
  const __bf16* brow0 = W + (size_t)(Nbase + (lane & 15)) * HIDD;
  const __bf16* brow1 = brow0 + 16 * HIDD;

  for (int k0 = 0; k0 < HIDD; k0 += 32) {
    v16bf a0 = ldA(arow0, k0, lane);
    v16bf a1 = ldA(arow1, k0, lane);
    v16bf fb0 = ldB(brow0, k0, lane);
    v16bf fb1 = ldB(brow1, k0, lane);
    acc[0][0] = wmma_bf16(a0, fb0, acc[0][0]);
    acc[0][1] = wmma_bf16(a0, fb1, acc[0][1]);
    acc[1][0] = wmma_bf16(a1, fb0, acc[1][0]);
    acc[1][1] = wmma_bf16(a1, fb1, acc[1][1]);
  }

#pragma unroll
  for (int mt = 0; mt < 2; ++mt)
#pragma unroll
    for (int nt = 0; nt < 2; ++nt)
#pragma unroll
      for (int r = 0; r < 8; ++r) {
        int m = Mbase + mt * 16 + r + ((lane >> 4) << 3);
        int n = Nbase + nt * 16 + (lane & 15);
        float v = acc[mt][nt][r] + bias[n];
        int bb = m >> 10, l = m & 1023;
        int h  = n >> 6,  d = n & 63;
        if (mode == 0)
          qws[((size_t)(bb * NHH + h) * LL + l) * HDD + d] = (__bf16)v;
        else if (mode == 1)
          kws[((size_t)(bb * NHH + h) * LL + l) * HDD + d] = (__bf16)v;
        else
          vtws[((size_t)(bb * NHH + h) * HDD + d) * LL + l] = (__bf16)v;
      }
}

// ---------------------------------------------------------------- fused scores + softmax + attn@V
// One block per (b, h, 16-query slab). 8 waves; wave w owns key columns [w*128, w*128+128).
// Softmax is register-resident: each wave's 8 score accumulators hold its 16x128 slice.
// Probs: f32 -> d_out (required output), bf16 -> LDS slab -> P@V WMMA A-fragments.
// 16x64 head output reduced across waves via LDS ds_add_f32 atomics.
__global__ void __launch_bounds__(256)
attn_fused_kernel(const __bf16* __restrict__ qws,
                  const __bf16* __restrict__ kws,
                  const __bf16* __restrict__ vtws,
                  const float* __restrict__ attn_bias,
                  const unsigned char* __restrict__ mask,
                  float* __restrict__ attnw,
                  __bf16* __restrict__ ctx) {
  __shared__ __bf16 pb[8][16 * PSTR];   // 34,816 B: per-wave bf16 prob slabs
  __shared__ float  redmax[16][8];
  __shared__ float  redsum[16][8];
  __shared__ float  oacc[16 * 64];      // 4 KB head-output accumulator

  const int lane = threadIdx.x & 31;
  const int wave = threadIdx.x >> 5;
  const int hi   = lane >> 4;   // 0/1
  const int lo   = lane & 15;
  const int qb = blockIdx.x * 16;
  const int h = blockIdx.y, b = blockIdx.z;

  for (int i = threadIdx.x; i < 16 * 64; i += 256) oacc[i] = 0.f;

  const size_t headoff = (size_t)(b * NHH + h);
  const __bf16* Q  = qws  + headoff * LL * HDD;
  const __bf16* Kp = kws  + headoff * LL * HDD;
  const __bf16* Vt = vtws + headoff * HDD * LL;

  // ---- scores: 16 x 128 per wave via 8 ntiles x 2 WMMA (HD = 64) ----
  const __bf16* qrow = Q + (size_t)(qb + lo) * HDD;
  v16bf a0 = ldA(qrow, 0, lane);
  v16bf a1 = ldA(qrow, 32, lane);

  v8f c[8];
#pragma unroll
  for (int t = 0; t < 8; ++t) {
    int nt = wave * 8 + t;
    const __bf16* krow = Kp + (size_t)(nt * 16 + lo) * HDD;
    v16bf fb0 = ldB(krow, 0, lane);
    v16bf fb1 = ldB(krow, 32, lane);
    v8f cc = {};
    cc = wmma_bf16(a0, fb0, cc);
    cc = wmma_bf16(a1, fb1, cc);
    c[t] = cc;
  }

  // ---- scale + bias + key-mask, in registers ----
  const float inv_scale = 0.125f;  // 1/sqrt(64)
  const float* biasbase = attn_bias + (headoff * LL + qb) * LL;
  const unsigned char* mrow = mask + (size_t)b * LL;
#pragma unroll
  for (int t = 0; t < 8; ++t) {
    int j = (wave * 8 + t) * 16 + lo;
    float mj = mrow[j] ? 0.f : -1e30f;
#pragma unroll
    for (int r = 0; r < 8; ++r) {
      int m = r + hi * 8;
      c[t][r] = c[t][r] * inv_scale + biasbase[(size_t)m * LL + j] + mj;
    }
  }

  // ---- row max: intra-wave (16-lane shfl) then cross-wave via LDS ----
#pragma unroll
  for (int r = 0; r < 8; ++r) {
    float v = c[0][r];
#pragma unroll
    for (int t = 1; t < 8; ++t) v = fmaxf(v, c[t][r]);
#pragma unroll
    for (int off = 8; off; off >>= 1) v = fmaxf(v, __shfl_xor(v, off, 32));
    if (lo == 0) redmax[r + hi * 8][wave] = v;
  }
  __syncthreads();
  float rmax[8];
#pragma unroll
  for (int r = 0; r < 8; ++r) {
    int m = r + hi * 8;
    float v = redmax[m][0];
#pragma unroll
    for (int w = 1; w < 8; ++w) v = fmaxf(v, redmax[m][w]);
    rmax[r] = v;
  }

  // ---- exp + row sum ----
#pragma unroll
  for (int r = 0; r < 8; ++r) {
    float s = 0.f;
#pragma unroll
    for (int t = 0; t < 8; ++t) {
      float e = __expf(c[t][r] - rmax[r]);
      c[t][r] = e;
      s += e;
    }
#pragma unroll
    for (int off = 8; off; off >>= 1) s += __shfl_xor(s, off, 32);
    if (lo == 0) redsum[r + hi * 8][wave] = s;
  }
  __syncthreads();
  float rinv[8];
#pragma unroll
  for (int r = 0; r < 8; ++r) {
    int m = r + hi * 8;
    float s = redsum[m][0];
#pragma unroll
    for (int w = 1; w < 8; ++w) s += redsum[m][w];
    rinv[r] = 1.f / s;
  }

  // ---- normalize: f32 probs -> global output, bf16 probs -> LDS slab ----
  float* prow0 = attnw + (headoff * LL + qb) * LL;
#pragma unroll
  for (int t = 0; t < 8; ++t) {
    int j  = (wave * 8 + t) * 16 + lo;  // global column
    int jj = t * 16 + lo;               // wave-local column
#pragma unroll
    for (int r = 0; r < 8; ++r) {
      int m = r + hi * 8;
      float p = c[t][r] * rinv[r];
      prow0[(size_t)m * LL + j] = p;
      pb[wave][m * PSTR + jj] = (__bf16)p;
    }
  }
  __syncthreads();

  // ---- P @ V over this wave's 128 key columns: 4 ksteps x 4 ntiles WMMA ----
  v8f acc[4] = {};
  const __bf16* pbw = &pb[wave][0];
  const int jbase = wave * 128;
  for (int st = 0; st < 4; ++st) {
    int jj0 = st * 32;
    const __bf16* pA = pbw + lo * PSTR + jj0 + (hi << 3);
    v16bf a = cat8(*(const v8bf*)pA, *(const v8bf*)(pA + 16));
#pragma unroll
    for (int nt = 0; nt < 4; ++nt) {
      const __bf16* vrow = Vt + (size_t)(nt * 16 + lo) * LL;
      v16bf bm = ldB(vrow, jbase + jj0, lane);
      acc[nt] = wmma_bf16(a, bm, acc[nt]);
    }
  }

  // ---- cross-wave K-partial reduction (ds_add_f32) ----
#pragma unroll
  for (int nt = 0; nt < 4; ++nt)
#pragma unroll
    for (int r = 0; r < 8; ++r) {
      int m = r + hi * 8;
      int d = nt * 16 + lo;
      atomicAdd(&oacc[m * 64 + d], acc[nt][r]);
    }
  __syncthreads();

  // ---- store bf16 ctx (head-concatenated [b, l, h, d]) ----
  for (int i = threadIdx.x; i < 16 * 64; i += 256) {
    int m = i >> 6, d = i & 63;
    ctx[(((size_t)b * LL + qb + m) * NHH + h) * HDD + d] = (__bf16)oacc[i];
  }
}

// ---------------------------------------------------------------- output projection
__global__ void out_proj_kernel(const __bf16* __restrict__ ctx,
                                const __bf16* __restrict__ wob,
                                const float* __restrict__ bo,
                                float* __restrict__ out) {
  const int lane = threadIdx.x & 31;
  const int wave = threadIdx.x >> 5;
  const int Mbase = blockIdx.x * 64 + (wave & 1) * 32;
  const int Nbase = blockIdx.y * 64 + (wave >> 1) * 32;

  v8f acc[2][2] = {};
  const __bf16* arow0 = ctx + (size_t)(Mbase + (lane & 15)) * HIDD;
  const __bf16* arow1 = arow0 + 16 * HIDD;
  const __bf16* brow0 = wob + (size_t)(Nbase + (lane & 15)) * HIDD;
  const __bf16* brow1 = brow0 + 16 * HIDD;

  for (int k0 = 0; k0 < HIDD; k0 += 32) {
    v16bf a0 = ldA(arow0, k0, lane);
    v16bf a1 = ldA(arow1, k0, lane);
    v16bf fb0 = ldB(brow0, k0, lane);
    v16bf fb1 = ldB(brow1, k0, lane);
    acc[0][0] = wmma_bf16(a0, fb0, acc[0][0]);
    acc[0][1] = wmma_bf16(a0, fb1, acc[0][1]);
    acc[1][0] = wmma_bf16(a1, fb0, acc[1][0]);
    acc[1][1] = wmma_bf16(a1, fb1, acc[1][1]);
  }

#pragma unroll
  for (int mt = 0; mt < 2; ++mt)
#pragma unroll
    for (int nt = 0; nt < 2; ++nt)
#pragma unroll
      for (int r = 0; r < 8; ++r) {
        int m = Mbase + mt * 16 + r + ((lane >> 4) << 3);
        int n = Nbase + nt * 16 + (lane & 15);
        out[(size_t)m * HIDD + n] = acc[mt][nt][r] + bo[n];
      }
}

// ---------------------------------------------------------------- launch
extern "C" void kernel_launch(void* const* d_in, const int* in_sizes, int n_in,
                              void* d_out, int out_size, void* d_ws, size_t ws_size,
                              hipStream_t stream) {
  const float* x         = (const float*)d_in[0];
  const float* attn_bias = (const float*)d_in[1];
  const unsigned char* mask = (const unsigned char*)d_in[2];
  const float* Wq = (const float*)d_in[3];
  const float* bq = (const float*)d_in[4];
  const float* Wk = (const float*)d_in[5];
  const float* bk = (const float*)d_in[6];
  const float* Wv = (const float*)d_in[7];
  const float* bv = (const float*)d_in[8];
  const float* Wo = (const float*)d_in[9];
  const float* bo = (const float*)d_in[10];

  float* out   = (float*)d_out;                 // [B, L, HID]
  float* attnw = out + (size_t)BB * LL * HIDD;  // [B, NH, L, L]

  // workspace layout (bf16 elements)
  __bf16* ws = (__bf16*)d_ws;
  const size_t nX = (size_t)BB * LL * HIDD;  // 8M
  const size_t nW = (size_t)HIDD * HIDD;     // 1M
  __bf16* xb   = ws;
  __bf16* wqb  = xb + nX;
  __bf16* wkb  = wqb + nW;
  __bf16* wvb  = wkb + nW;
  __bf16* wob  = wvb + nW;
  __bf16* qws  = wob + nW;   // [b,h,l,d]
  __bf16* kws  = qws + nX;   // [b,h,l,d]
  __bf16* vtws = kws + nX;   // [b,h,d,l]
  __bf16* ctx  = vtws + nX;  // [b,l,h,d]
  (void)ws_size; (void)in_sizes; (void)n_in; (void)out_size;

  // 1) f32 -> bf16 conversions
  cvt_bf16_kernel<<<(int)(nX / 4 / 256), 256, 0, stream>>>(x,  xb,  (int)nX);
  cvt_bf16_kernel<<<(int)(nW / 4 / 256), 256, 0, stream>>>(Wq, wqb, (int)nW);
  cvt_bf16_kernel<<<(int)(nW / 4 / 256), 256, 0, stream>>>(Wk, wkb, (int)nW);
  cvt_bf16_kernel<<<(int)(nW / 4 / 256), 256, 0, stream>>>(Wv, wvb, (int)nW);
  cvt_bf16_kernel<<<(int)(nW / 4 / 256), 256, 0, stream>>>(Wo, wob, (int)nW);

  // 2) QKV projections (M=8192, N=1024 tiled 64x64 per block, z = which matrix)
  qkv_gemm_kernel<<<dim3(128, 16, 3), 128, 0, stream>>>(xb, wqb, wkb, wvb, bq, bk, bv,
                                                        qws, kws, vtws);

  // 3) fused scores + softmax + attn@V  (probs -> d_out, ctx -> ws)
  attn_fused_kernel<<<dim3(LL / 16, NHH, BB), 256, 0, stream>>>(qws, kws, vtws, attn_bias,
                                                                mask, attnw, ctx);

  // 4) output projection -> out (f32, start of d_out)
  out_proj_kernel<<<dim3(128, 16), 128, 0, stream>>>(ctx, wob, bo, out);
}
